// LSTM_50019189129878
// MI455X (gfx1250) — compile-verified, matrix operand order checked by reading, not run on hardware
//
#include <hip/hip_runtime.h>

typedef __attribute__((ext_vector_type(16))) _Float16 v16h;
typedef __attribute__((ext_vector_type(8)))  float    v8f;
typedef __attribute__((ext_vector_type(4)))  _Float16 v4h;

constexpr int T_SEQ = 512;
constexpr int I_IN  = 64;
constexpr int HID   = 128;
constexpr int GW    = 4 * HID;           // 512 gate columns
constexpr int BM    = 16;                // batch rows per block

// packed f16 element counts in workspace
constexpr int WIH_ELTS = 32 * 2 * 32 * 16;   // 32768
constexpr int WHH_ELTS = 32 * 4 * 32 * 16;   // 65536
constexpr int SX_ELTS  = 2 * 32 * 16;        // 1024 (one x A-tile set)
constexpr int SH_ELTS  = 4 * 32 * 16;        // 2048 (one h A-tile set)

// --- WMMA f16 operand layout helpers (ISA 7.12.2, 16-bit 16x32 A / 32x16 B) ---
__device__ __host__ inline int kl_from_lane_j(int lane, int j) {
    int hi = lane >> 4;
    int v  = j >> 1;
    int p  = j & 1;
    int base = (v < 4) ? (v * 2 + p) : (16 + (v - 4) * 2 + p);
    return base + hi * 8;
}

// CDNA5 hardware tanh (TRANS32 op) — branch-free activation path.
__device__ inline float tanh_hw(float v) {
    float r;
    asm("v_tanh_f32 %0, %1" : "=v"(r) : "v"(v));
    return r;
}
// sigmoid(x) = 0.5*tanh(x/2) + 0.5  (1 trans + 2 VALU, no branches)
__device__ inline float sig_hw(float v) {
    return __builtin_fmaf(0.5f, tanh_hw(0.5f * v), 0.5f);
}

// ----------------------------------------------------------------------------
// Prep: pack W_ih^T, W_hh^T into f16 WMMA B-tile layout; fuse biases.
// ----------------------------------------------------------------------------
__global__ void lstm_prep(const float* __restrict__ Wih, const float* __restrict__ Whh,
                          const float* __restrict__ bih, const float* __restrict__ bhh,
                          _Float16* __restrict__ wih_p, _Float16* __restrict__ whh_p,
                          float* __restrict__ bias_p) {
    int idx = blockIdx.x * blockDim.x + threadIdx.x;
    if (idx < WIH_ELTS) {
        int nt   = idx >> 10;
        int kc   = (idx >> 9) & 1;
        int lane = (idx >> 4) & 31;
        int j    = idx & 15;
        int K    = kc * 32 + kl_from_lane_j(lane, j);
        int N    = nt * 16 + (lane & 15);
        wih_p[idx] = (_Float16)Wih[N * I_IN + K];          // B[k][n] = W_ih[n][k]
    } else if (idx < WIH_ELTS + WHH_ELTS) {
        int i2   = idx - WIH_ELTS;
        int nt   = i2 >> 11;
        int kc   = (i2 >> 9) & 3;
        int lane = (i2 >> 4) & 31;
        int j    = i2 & 15;
        int K    = kc * 32 + kl_from_lane_j(lane, j);
        int N    = nt * 16 + (lane & 15);
        whh_p[i2] = (_Float16)Whh[N * HID + K];
    } else if (idx < WIH_ELTS + WHH_ELTS + GW) {
        int n = idx - (WIH_ELTS + WHH_ELTS);
        bias_p[n] = bih[n] + bhh[n];
    }
}

// ----------------------------------------------------------------------------
// Main persistent LSTM kernel: 1 block = 16 batch rows, 8 waves.
// ALL weight B-tiles + bias C-vectors live in VGPRs (loop-invariant).
// c-state in VGPRs, h-state double-buffered in LDS (f16, A-tile layout).
// One barrier per timestep; branch-free v_tanh_f32 activations.
// ----------------------------------------------------------------------------
__global__ __launch_bounds__(256, 1)
void lstm_main(const float* __restrict__ x, const _Float16* __restrict__ wih_p,
               const _Float16* __restrict__ whh_p, const float* __restrict__ bias_p,
               float* __restrict__ out) {
    __shared__ __align__(64) _Float16 sX[2][SX_ELTS];   // 2 x 2KB
    __shared__ __align__(64) _Float16 sH[2][SH_ELTS];   // 2 x 4KB

    const int tid  = threadIdx.x;
    const int lane = tid & 31;
    const int w    = tid >> 5;                      // wave 0..7 = hidden tile
    const int b0   = blockIdx.x * BM;

    const int laneN = lane & 15;
    const int hiL   = lane >> 4;
    const int col   = w * 16 + laneN;               // hidden column owned by this lane

    // ---- hoist ALL weight B-tiles into registers (loop-invariant) ----
    v16h bIH[2][4];                                 // [kc][gate]
    v16h bHH[4][4];                                 // [kc][gate]
#pragma unroll
    for (int kc = 0; kc < 2; ++kc)
#pragma unroll
        for (int g = 0; g < 4; ++g)
            bIH[kc][g] = *(const v16h*)(wih_p + (((w + g * 8) * 2 + kc) * 32 + lane) * 16);
#pragma unroll
    for (int kc = 0; kc < 4; ++kc)
#pragma unroll
        for (int g = 0; g < 4; ++g)
            bHH[kc][g] = *(const v16h*)(whh_p + (((w + g * 8) * 4 + kc) * 32 + lane) * 16);

    // ---- loop-invariant bias C-vectors (seed the first WMMA of each chain) ----
    v8f biasv[4];
#pragma unroll
    for (int g = 0; g < 4; ++g) {
        float b = bias_p[g * HID + col];
#pragma unroll
        for (int r = 0; r < 8; ++r) biasv[g][r] = b;
    }

    float cst[8];
#pragma unroll
    for (int r = 0; r < 8; ++r) cst[r] = 0.0f;

    // x staging role: thread -> (row m, 4 input channels), A-tile packed slot
    const int xm  = tid >> 4;
    const int xc  = (tid & 15) << 2;
    const int xkc = xc >> 5;
    const int xkl = xc & 31;
    const int xhi = (xkl >> 3) & 1;
    const int xj  = (xkl & 7) + ((xkl >> 4) << 3);
    const int xlds = ((xkc * 32) + (xm + 16 * xhi)) * 16 + xj;
    const float* xrow = x + ((size_t)(b0 + xm) * T_SEQ) * I_IN + xc;

    // h writeback slot pieces (loop-invariant): hidden index = col
    const int hkc = col >> 5;
    const int hkl = col & 31;
    const int hhi = (hkl >> 3) & 1;
    const int hj  = (hkl & 7) + ((hkl >> 4) << 3);

    // ---- prologue: x_0 into buffer 0, zero h buffer 0 ----
    {
        float4 xv = *(const float4*)(xrow);
        v4h hx = { (_Float16)xv.x, (_Float16)xv.y, (_Float16)xv.z, (_Float16)xv.w };
        *(v4h*)(&sX[0][xlds]) = hx;
        uint* zh = (uint*)&sH[0][0];
        for (int i = tid; i < SH_ELTS / 2; i += 256) zh[i] = 0u;
    }

    for (int t = 0; t < T_SEQ; ++t) {
        const int p = t & 1;
        const int q = p ^ 1;
        __syncthreads();   // buf[p] (x_t, h_t-1) ready; all reads of buf[q] finished

        // ---- stage x_{t+1} into buffer q early (hide global latency) ----
        if (t + 1 < T_SEQ) {
            float4 xv = *(const float4*)(xrow + (size_t)(t + 1) * I_IN);
            v4h hx = { (_Float16)xv.x, (_Float16)xv.y, (_Float16)xv.z, (_Float16)xv.w };
            *(v4h*)(&sX[q][xlds]) = hx;
            if (t + 2 < T_SEQ) __builtin_prefetch(xrow + (size_t)(t + 2) * I_IN, 0, 3);
        }

        // ---- preload all A-tiles for this step (overlap DS latency w/ WMMA) ----
        v16h aX[2], aH[4];
#pragma unroll
        for (int kc = 0; kc < 2; ++kc)
            aX[kc] = *(const v16h*)(&sX[p][(kc * 32 + lane) * 16]);
#pragma unroll
        for (int kc = 0; kc < 4; ++kc)
            aH[kc] = *(const v16h*)(&sH[p][(kc * 32 + lane) * 16]);

        // ---- gates = bias + x_t@Wih^T + h_{t-1}@Whh^T (24 WMMAs, 4 chains) ----
        v8f acc[4];
#pragma unroll
        for (int g = 0; g < 4; ++g)   // first k-step seeds from bias C-vector
            acc[g] = __builtin_amdgcn_wmma_f32_16x16x32_f16(
                false, aX[0], false, bIH[0][g], (short)0, biasv[g], false, false);
#pragma unroll
        for (int g = 0; g < 4; ++g)
            acc[g] = __builtin_amdgcn_wmma_f32_16x16x32_f16(
                false, aX[1], false, bIH[1][g], (short)0, acc[g], false, false);
#pragma unroll
        for (int kc = 0; kc < 4; ++kc)
#pragma unroll
            for (int g = 0; g < 4; ++g)
                acc[g] = __builtin_amdgcn_wmma_f32_16x16x32_f16(
                    false, aH[kc], false, bHH[kc][g], (short)0, acc[g], false, false);

        // ---- elementwise cell update: branch-free v_tanh_f32 activations ----
        float hval[8];
#pragma unroll
        for (int r = 0; r < 8; ++r) {
            float ig = sig_hw(acc[0][r]);
            float fg = sig_hw(acc[1][r]);
            float gg = tanh_hw(acc[2][r]);
            float og = sig_hw(acc[3][r]);
            float c  = fg * cst[r] + ig * gg;
            cst[r]   = c;
            hval[r]  = og * tanh_hw(c);
        }

        // ---- write h_t: f16 -> sH[q] (A-tile layout), fp32 -> out (streamed) ----
#pragma unroll
        for (int r = 0; r < 8; ++r) {
            int M = r + 8 * hiL;
            sH[q][((hkc * 32) + (M + 16 * hhi)) * 16 + hj] = (_Float16)hval[r];
            __builtin_nontemporal_store(hval[r],
                &out[((size_t)(b0 + M) * T_SEQ + t) * HID + col]);
        }
    }
}

// ----------------------------------------------------------------------------
extern "C" void kernel_launch(void* const* d_in, const int* in_sizes, int n_in,
                              void* d_out, int out_size, void* d_ws, size_t ws_size,
                              hipStream_t stream) {
    const float* x   = (const float*)d_in[0];
    const float* Wih = (const float*)d_in[1];
    const float* Whh = (const float*)d_in[2];
    const float* bih = (const float*)d_in[3];
    const float* bhh = (const float*)d_in[4];
    float*       out = (float*)d_out;

    _Float16* wih_p  = (_Float16*)d_ws;
    _Float16* whh_p  = wih_p + WIH_ELTS;
    float*    bias_p = (float*)((char*)d_ws + (size_t)(WIH_ELTS + WHH_ELTS) * 2);

    {
        int total  = WIH_ELTS + WHH_ELTS + GW;
        int blocks = (total + 255) / 256;
        lstm_prep<<<blocks, 256, 0, stream>>>(Wih, Whh, bih, bhh, wih_p, whh_p, bias_p);
    }

    lstm_main<<<2048 / BM, 256, 0, stream>>>(x, wih_p, whh_p, bias_p, out);
}
